// NFM_6811818132385
// MI455X (gfx1250) — matrix operand order repaired and checked by reading, not compiled, not run on hardware
//
#include <hip/hip_runtime.h>
#include <hip/hip_bf16.h>
#include <math.h>

typedef __attribute__((ext_vector_type(16))) _Float16 v16h;
typedef __attribute__((ext_vector_type(8)))  _Float16 v8h;
typedef __attribute__((ext_vector_type(8)))  float    v8f;

#define B_SAMPLES 16384
#define N_SPARSE  26
#define N_DENSE   13
#define VOCAB     100000
#define EMB_D     16
#define XCOLS     32          // 29 padded to 32
#define H1N       256
#define H2N       128
#define H3N       64
#define BN_EPS    1e-5f
#define LDS_PAD   8           // +8 halves = +16B per row -> conflict-free strided reads

// ---------------- workspace layout (bytes) ----------------
#define OFF_XRAW   ((size_t)0)                                   // B*32 f32  = 2 MB
#define OFF_PART   (OFF_XRAW + (size_t)B_SAMPLES*XCOLS*4)        // 1024*64 f32 = 256 KB
#define OFF_SS     (OFF_PART + (size_t)1024*64*4)                // 64 f32
#define OFF_W1P    (OFF_SS   + (size_t)256)                      // 32*256 f16
#define OFF_W2P    (OFF_W1P  + (size_t)32*256*2)                 // 256*128 f16
#define OFF_W3P    (OFF_W2P  + (size_t)256*128*2)                // 128*64 f16

// Swizzle one element of a [K x N] f32 weight into the wave32 WMMA B-fragment
// layout: dst[ ((kt*(N/16) + nT)*32 + lane)*16 + e ], lane = ((k%32)/8%2)*16 + n%16, e = (k%32)/16*8 + k%8
__device__ __forceinline__ void prep_one(const float* __restrict__ W,
                                         _Float16* __restrict__ dst,
                                         int N, int Ksrc, int i) {
    int k = i / N, n = i % N;
    int kt = k >> 5, kk = k & 31, gg = (kk >> 3) & 1, e = ((kk >> 4) << 3) + (kk & 7);
    int nT = n >> 4, nn = n & 15, lane = gg * 16 + nn;
    float v = (k < Ksrc) ? W[k * N + n] : 0.0f;
    dst[(((size_t)kt * (N >> 4) + nT) * 32 + lane) * 16 + e] = (_Float16)v;
}

__global__ void nfm_wprep_all(const float* __restrict__ W1, const float* __restrict__ W2,
                              const float* __restrict__ W3,
                              _Float16* __restrict__ W1p, _Float16* __restrict__ W2p,
                              _Float16* __restrict__ W3p) {
    int i = blockIdx.x * 256 + threadIdx.x;                // 0 .. 49151
    if (i < 32 * 256)                 prep_one(W1, W1p, 256, 29,  i);
    else if (i < 32 * 256 + 256*128)  prep_one(W2, W2p, 128, 256, i - 32 * 256);
    else                              prep_one(W3, W3p, 64,  128, i - 32 * 256 - 256 * 128);
}

// FM pooling + concat + per-block BN partial sums. 256 threads = 16 samples x 16 dims.
__global__ void nfm_fm_pool(const float* __restrict__ xd, const int* __restrict__ xs,
                            const float* __restrict__ emb,
                            float* __restrict__ Xraw, float* __restrict__ partials) {
    int t  = threadIdx.x;
    int sl = t >> 4;                  // local sample 0..15
    int d  = t & 15;                  // embedding dim 0..15
    int s  = blockIdx.x * 16 + sl;

    const int* idx = xs + (size_t)s * N_SPARSE;
    float sum = 0.0f, sq = 0.0f;
    #pragma unroll 1
    for (int f = 0; f < N_SPARSE; ++f) {
        int id  = idx[f];
        float e = emb[((size_t)f * VOCAB + (size_t)id) * EMB_D + d];
        sum += e;
        sq  += e * e;
    }
    float cross = 0.5f * (sum * sum - sq);
    float dense = (d < N_DENSE) ? xd[(size_t)s * N_DENSE + d] : 0.0f;

    float* xr = Xraw + (size_t)s * XCOLS;
    xr[d]      = cross;               // cols 0..15
    xr[16 + d] = dense;               // cols 16..28 real, 29..31 zero

    __shared__ float sh[16][XCOLS];
    sh[sl][d]      = cross;
    sh[sl][16 + d] = dense;
    __syncthreads();

    if (t < XCOLS) {                  // deterministic fixed-order reduction
        float ps = 0.0f, pq = 0.0f;
        #pragma unroll
        for (int i = 0; i < 16; ++i) {
            float v = sh[i][t];
            ps += v;
            pq += v * v;
        }
        partials[blockIdx.x * 64 + t]      = ps;
        partials[blockIdx.x * 64 + 32 + t] = pq;
    }
}

// Reduce 1024 partials -> fused BN scale/shift per column (pad cols -> 0).
// 512 threads: 8 chunks x 64 slots, two-level deterministic reduction.
__global__ void nfm_bn_stats(const float* __restrict__ partials,
                             const float* __restrict__ gamma,
                             const float* __restrict__ beta,
                             float* __restrict__ ss) {
    __shared__ float red[8][64];
    int t  = threadIdx.x;             // 0..511
    int c  = t & 63;                  // slot: 0..31 sum, 32..63 sumsq
    int ch = t >> 6;                  // chunk 0..7
    float acc = 0.0f;
    #pragma unroll 1
    for (int b = ch * 128; b < ch * 128 + 128; ++b) acc += partials[b * 64 + c];
    red[ch][c] = acc;
    __syncthreads();
    if (t < 64) {
        float s = 0.0f;
        #pragma unroll
        for (int i = 0; i < 8; ++i) s += red[i][t];
        red[0][t] = s;
    }
    __syncthreads();
    if (t < 32) {
        float s = red[0][t], q = red[0][32 + t];
        float mean = s * (1.0f / (float)B_SAMPLES);
        float var  = q * (1.0f / (float)B_SAMPLES) - mean * mean;
        float sc = 0.0f, sf = 0.0f;
        if (t < 29) {
            sc = gamma[t] * rsqrtf(var + BN_EPS);
            sf = beta[t] - mean * sc;
        }
        ss[t]      = sc;
        ss[32 + t] = sf;
    }
}

// Fully fused MLP: BN-fold + L1(29->256) + L2(256->128) + L3(128->64) + dot(64->1) + sigmoid.
// One block = 16 waves = one 16-sample strip. All intermediates live in LDS.
__global__ void __launch_bounds__(512) nfm_mlp_fused(
    const float* __restrict__ Xraw, const float* __restrict__ ss,
    const _Float16* __restrict__ W1p, const _Float16* __restrict__ W2p,
    const _Float16* __restrict__ W3p,
    const float* __restrict__ b1, const float* __restrict__ b2,
    const float* __restrict__ b3,
    const float* __restrict__ Wf, const float* __restrict__ bf,
    float* __restrict__ out) {
    __shared__ alignas(16) _Float16 H1s[16][H1N + LDS_PAD];  // stride 528B: 16B rotate/row
    __shared__ alignas(16) _Float16 H2s[16][H2N + LDS_PAD];  // stride 272B
    __shared__ alignas(16) _Float16 H3s[16][H3N + LDS_PAD];  // stride 144B

    int t    = threadIdx.x;
    int lane = t & 31;
    int wave = t >> 5;               // 0..15
    int g    = lane >> 4;
    int mloc = lane & 15;
    int tM   = blockIdx.x;
    int mrow = tM * 16 + mloc;

    // ---- Layer 1: wave w computes N-tile w (K=32, one WMMA) ----
    {
        const float* xr = Xraw + (size_t)mrow * XCOLS;
        const float* sc = ss;
        const float* sf = ss + 32;
        v16h a;
        #pragma unroll
        for (int e = 0; e < 16; ++e) {
            int k = ((e >> 3) << 4) + (g << 3) + (e & 7);    // ISA 16-bit A layout
            a[e] = (_Float16)(xr[k] * sc[k] + sf[k]);        // BN folded into A load
        }
        v16h b = *(const v16h*)(W1p + ((size_t)wave * 32 + lane) * 16);
        v8f c = {};
        c = __builtin_amdgcn_wmma_f32_16x16x32_f16(false, a, false, b, (short)0, c, false, false);
        int n = wave * 16 + mloc;
        float bias = b1[n];
        #pragma unroll
        for (int r = 0; r < 8; ++r) {                        // ISA f32 C/D layout
            float v = c[r] + bias;
            v = v > 0.0f ? v : 0.0f;
            H1s[r + 8 * g][n] = (_Float16)v;
        }
    }
    __syncthreads();

    // ---- Layer 2: waves 0..7, N-tile = wave, K=256 (8 WMMAs) ----
    if (wave < 8) {
        v8f c = {};
        #pragma unroll
        for (int kt = 0; kt < 8; ++kt) {
            v8h lo = *(const v8h*)(&H1s[mloc][kt * 32 + g * 8]);
            v8h hi = *(const v8h*)(&H1s[mloc][kt * 32 + 16 + g * 8]);
            v16h a;
            #pragma unroll
            for (int e = 0; e < 8; ++e) { a[e] = lo[e]; a[8 + e] = hi[e]; }
            v16h b = *(const v16h*)(W2p + (((size_t)kt * 8 + wave) * 32 + lane) * 16);
            c = __builtin_amdgcn_wmma_f32_16x16x32_f16(false, a, false, b, (short)0, c, false, false);
        }
        int n = wave * 16 + mloc;
        float bias = b2[n];
        #pragma unroll
        for (int r = 0; r < 8; ++r) {
            float v = c[r] + bias;
            v = v > 0.0f ? v : 0.0f;
            H2s[r + 8 * g][n] = (_Float16)v;
        }
    }
    __syncthreads();

    // ---- Layer 3: waves 0..3, N-tile = wave, K=128 (4 WMMAs) ----
    if (wave < 4) {
        v8f c = {};
        #pragma unroll
        for (int kt = 0; kt < 4; ++kt) {
            v8h lo = *(const v8h*)(&H2s[mloc][kt * 32 + g * 8]);
            v8h hi = *(const v8h*)(&H2s[mloc][kt * 32 + 16 + g * 8]);
            v16h a;
            #pragma unroll
            for (int e = 0; e < 8; ++e) { a[e] = lo[e]; a[8 + e] = hi[e]; }
            v16h b = *(const v16h*)(W3p + (((size_t)kt * 4 + wave) * 32 + lane) * 16);
            c = __builtin_amdgcn_wmma_f32_16x16x32_f16(false, a, false, b, (short)0, c, false, false);
        }
        int n = wave * 16 + mloc;
        float bias = b3[n];
        #pragma unroll
        for (int r = 0; r < 8; ++r) {
            float v = c[r] + bias;
            v = v > 0.0f ? v : 0.0f;
            H3s[r + 8 * g][n] = (_Float16)v;
        }
    }
    __syncthreads();

    // ---- Final layer + sigmoid: one thread per sample ----
    if (t < 16) {
        float acc = bf[0];
        #pragma unroll
        for (int j = 0; j < H3N; ++j) acc += (float)H3s[t][j] * Wf[j];
        out[tM * 16 + t] = 1.0f / (1.0f + __expf(-acc));
    }
}

extern "C" void kernel_launch(void* const* d_in, const int* in_sizes, int n_in,
                              void* d_out, int out_size, void* d_ws, size_t ws_size,
                              hipStream_t stream) {
    const float* x_dense  = (const float*)d_in[0];
    const int*   x_sparse = (const int*)  d_in[1];
    const float* emb      = (const float*)d_in[2];
    const float* gamma    = (const float*)d_in[3];
    const float* beta     = (const float*)d_in[4];
    const float* W1       = (const float*)d_in[5];
    const float* b1       = (const float*)d_in[6];
    const float* W2       = (const float*)d_in[7];
    const float* b2       = (const float*)d_in[8];
    const float* W3       = (const float*)d_in[9];
    const float* b3       = (const float*)d_in[10];
    const float* Wf       = (const float*)d_in[11];
    const float* bf       = (const float*)d_in[12];
    float* out = (float*)d_out;

    char* ws = (char*)d_ws;
    float*    Xraw = (float*)   (ws + OFF_XRAW);
    float*    part = (float*)   (ws + OFF_PART);
    float*    ss   = (float*)   (ws + OFF_SS);
    _Float16* W1p  = (_Float16*)(ws + OFF_W1P);
    _Float16* W2p  = (_Float16*)(ws + OFF_W2P);
    _Float16* W3p  = (_Float16*)(ws + OFF_W3P);

    // 1) Weight conversion + B-fragment swizzle (49152 elements)
    nfm_wprep_all<<<192, 256, 0, stream>>>(W1, W2, W3, W1p, W2p, W3p);

    // 2) Embedding gather + FM pooling + BN partials
    nfm_fm_pool<<<B_SAMPLES / 16, 256, 0, stream>>>(x_dense, x_sparse, emb, Xraw, part);

    // 3) BN statistics -> fused scale/shift
    nfm_bn_stats<<<1, 512, 0, stream>>>(part, gamma, beta, ss);

    // 4) Fully fused WMMA MLP + sigmoid
    nfm_mlp_fused<<<B_SAMPLES / 16, 512, 0, stream>>>(Xraw, ss, W1p, W2p, W3p,
                                                      b1, b2, b3, Wf, bf, out);
}
